// LEConv_936302871074
// MI455X (gfx1250) — matrix operand order repaired and checked by reading, not compiled, not run on hardware
//
#include <hip/hip_runtime.h>

#define D 256  // feature dim (D_IN == D_OUT == 256)

typedef __attribute__((ext_vector_type(16))) __bf16 v16bf;
typedef __attribute__((ext_vector_type(8)))  float  v8f;

// ---------- helpers ----------
__device__ __forceinline__ unsigned short f2bf(float f) {
  // round-to-nearest-even f32 -> bf16
  unsigned u = __builtin_bit_cast(unsigned, f);
  u += 0x7fffu + ((u >> 16) & 1u);
  return (unsigned short)(u >> 16);
}

// ---------- prep: x (f32) -> xh (bf16), vectorized x4 ----------
__global__ __launch_bounds__(256) void prep_x(const float* __restrict__ x,
                                              unsigned short* __restrict__ xh,
                                              long long n4) {
  long long t = (long long)blockIdx.x * blockDim.x + threadIdx.x;
  if (t >= n4) return;
  const float4 v = ((const float4*)x)[t];
  ushort4 h;
  h.x = f2bf(v.x); h.y = f2bf(v.y); h.z = f2bf(v.z); h.w = f2bf(v.w);
  ((ushort4*)xh)[t] = h;
}

// ---------- prep: W[k][n] f32 -> WT[w][n][k] bf16 (transposed for B frags) ----------
__global__ __launch_bounds__(256) void prep_w(const float* __restrict__ w0,
                                              const float* __restrict__ w1,
                                              const float* __restrict__ w2,
                                              unsigned short* __restrict__ wt) {
  int t = blockIdx.x * blockDim.x + threadIdx.x;   // t < 3*D*D
  if (t >= 3 * D * D) return;
  int w = t >> 16;            // D*D == 65536
  int n = (t >> 8) & (D - 1);
  int k = t & (D - 1);
  const float* src = (w == 0) ? w0 : (w == 1) ? w1 : w2;
  wt[(size_t)w * D * D + (size_t)n * D + k] = f2bf(src[(size_t)k * D + n]);
}

// ---------- zero degree array ----------
__global__ __launch_bounds__(256) void zero_deg(float* __restrict__ deg, int n) {
  int t = blockIdx.x * blockDim.x + threadIdx.x;
  if (t < n) deg[t] = 0.0f;
}

// ---------- degree histogram over col = edge_index[1] ----------
__global__ __launch_bounds__(256) void deg_hist(const long long* __restrict__ ei,
                                                float* __restrict__ deg, int E) {
  int e = blockIdx.x * blockDim.x + threadIdx.x;
  if (e < E) atomicAdd(&deg[(int)ei[(size_t)E + e]], 1.0f);
}

// ---------- fused triple GEMM + epilogue ----------
// One wave computes one 16x16 output tile for all three projections.
//   srcbuf = x@W_src + b_src                       (kept for edge gather)
//   out    = x@W_self + b_self - deg .* (x@W_dst + b_dst)   (agg base)
// N (=50000) is a multiple of 16, so no row tail handling needed.
__global__ __launch_bounds__(256) void leconv_gemm(
    const unsigned short* __restrict__ xh,      // [N][D] bf16
    const unsigned short* __restrict__ wtS,     // [D n][D k] bf16
    const unsigned short* __restrict__ wtD,
    const unsigned short* __restrict__ wtE,
    const float* __restrict__ bS, const float* __restrict__ bD,
    const float* __restrict__ bE, const float* __restrict__ deg,
    float* __restrict__ srcbuf, float* __restrict__ out, int nRowTiles) {
  int wave = (int)((blockIdx.x * blockDim.x + threadIdx.x) >> 5);
  int lane = threadIdx.x & 31;
  int rowTile = wave >> 4;      // 16 col tiles (D/16)
  int colTile = wave & 15;
  if (rowTile >= nRowTiles) return;   // uniform per wave -> EXEC stays all-1
  int row0 = rowTile * 16;
  int col0 = colTile * 16;
  int mn = lane & 15;           // A: row M;  B: col N;  C/D: col N
  int kg = lane >> 4;           // K-half selector

  v8f accS = {}, accD = {}, accE = {};
  const unsigned short* aB = xh + (size_t)(row0 + mn) * D;
  const unsigned short* sB = wtS + (size_t)(col0 + mn) * D;
  const unsigned short* dB = wtD + (size_t)(col0 + mn) * D;
  const unsigned short* eB = wtE + (size_t)(col0 + mn) * D;

  for (int k0 = 0; k0 < D; k0 += 32) {
    union { v16bf v; unsigned u[8]; } a, bs, bd, be;
#pragma unroll
    for (int i = 0; i < 8; ++i) {
      // A 16x32 bf16 layout: VGPR i holds K pair; lanes16-31 offset K by 8/8
      int ka = ((i < 4) ? 0 : 16) + kg * 8 + 2 * (i & 3);
      a.u[i] = *(const unsigned*)(aB + k0 + ka);
      // B 32x16 bf16 layout: same K order as A (VGPR i holds K pair; lanes16-31 offset K by 8/8) for col N=lane&15
      int kb = ((i < 4) ? 0 : 16) + kg * 8 + 2 * (i & 3);
      bs.u[i] = *(const unsigned*)(sB + k0 + kb);
      bd.u[i] = *(const unsigned*)(dB + k0 + kb);
      be.u[i] = *(const unsigned*)(eB + k0 + kb);
    }
    accS = __builtin_amdgcn_wmma_f32_16x16x32_bf16(false, a.v, false, bs.v,
                                                   (short)0, accS, false, false);
    accD = __builtin_amdgcn_wmma_f32_16x16x32_bf16(false, a.v, false, bd.v,
                                                   (short)0, accD, false, false);
    accE = __builtin_amdgcn_wmma_f32_16x16x32_bf16(false, a.v, false, be.v,
                                                   (short)0, accE, false, false);
  }

  // epilogue: C/D layout -> lane has column n = mn, rows m = kg*8 + r
  float biasS = bS[col0 + mn];
  float biasD = bD[col0 + mn];
  float biasE = bE[col0 + mn];
#pragma unroll
  for (int r = 0; r < 8; ++r) {
    int row = row0 + kg * 8 + r;
    size_t o = (size_t)row * D + col0 + mn;
    float sv = accS[r] + biasS;
    srcbuf[o] = sv;
    float dv = accD[r] + biasD;
    float ev = accE[r] + biasE;
    out[o] = ev - deg[row] * dv;   // self + b_self - deg*dst
  }
}

// ---------- edge scatter: out[col] += srcbuf[row], 64 threads/edge ----------
__global__ __launch_bounds__(256) void edge_scatter(const long long* __restrict__ ei,
                                                    const float* __restrict__ srcbuf,
                                                    float* __restrict__ out,
                                                    int E) {
  long long t = (long long)blockIdx.x * blockDim.x + threadIdx.x;
  int e = (int)(t >> 6);
  int q = (int)(t & 63);          // quad index: 64 * float4 = 256 feats
  if (e >= E) return;
  int r = (int)ei[e];
  int c = (int)ei[(size_t)E + e];
  const float4 v = *(const float4*)(srcbuf + (size_t)r * D + q * 4);
  float* o = out + (size_t)c * D + q * 4;
  atomicAdd(o + 0, v.x);
  atomicAdd(o + 1, v.y);
  atomicAdd(o + 2, v.z);
  atomicAdd(o + 3, v.w);
}

extern "C" void kernel_launch(void* const* d_in, const int* in_sizes, int n_in,
                              void* d_out, int out_size, void* d_ws, size_t ws_size,
                              hipStream_t stream) {
  const float*     x     = (const float*)d_in[0];
  const long long* ei    = (const long long*)d_in[1];
  const float*     Wsrc  = (const float*)d_in[2];
  const float*     bsrc  = (const float*)d_in[3];
  const float*     Wdst  = (const float*)d_in[4];
  const float*     bdst  = (const float*)d_in[5];
  const float*     Wself = (const float*)d_in[6];
  const float*     bself = (const float*)d_in[7];
  float*           out   = (float*)d_out;

  const int N = in_sizes[0] / D;   // 50000 (multiple of 16)
  const int E = in_sizes[1] / 2;   // 800000

  // workspace carve-out (~77.5 MB total)
  char* ws = (char*)d_ws;
  size_t off = 0;
  auto carve = [&](size_t bytes) -> void* {
    void* p = ws + off;
    off += (bytes + 255) & ~(size_t)255;
    return p;
  };
  unsigned short* xh     = (unsigned short*)carve((size_t)N * D * 2);
  unsigned short* wt     = (unsigned short*)carve((size_t)3 * D * D * 2);
  float*          srcbuf = (float*)carve((size_t)N * D * 4);
  float*          deg    = (float*)carve((size_t)N * 4);

  // 1. convert x -> bf16
  long long n4 = (long long)N * D / 4;
  prep_x<<<(unsigned)((n4 + 255) / 256), 256, 0, stream>>>(x, xh, n4);
  // 2. transpose+convert the three weight matrices
  prep_w<<<(3 * D * D + 255) / 256, 256, 0, stream>>>(Wsrc, Wdst, Wself, wt);
  // 3. degree histogram over destination nodes
  zero_deg<<<(N + 255) / 256, 256, 0, stream>>>(deg, N);
  deg_hist<<<(E + 255) / 256, 256, 0, stream>>>(ei, deg, E);
  // 4. fused triple GEMM (bf16 WMMA) + epilogue writing out-init & srcbuf
  int nRowTiles = N / 16;
  long long waves = (long long)nRowTiles * (D / 16);
  leconv_gemm<<<(unsigned)((waves + 7) / 8), 256, 0, stream>>>(
      xh, wt, wt + (size_t)D * D, wt + (size_t)2 * D * D,
      bsrc, bdst, bself, deg, srcbuf, out, nRowTiles);
  // 5. edge scatter: out[col] += src[row]  (L2-resident atomics)
  long long st = (long long)E * 64;
  edge_scatter<<<(unsigned)((st + 255) / 256), 256, 0, stream>>>(ei, srcbuf, out, E);
}
